// MultilayerGRU_6846177869903
// MI455X (gfx1250) — matrix operand order repaired and checked by reading, not compiled, not run on hardware
//
#include <hip/hip_runtime.h>
#include <hip/hip_bf16.h>

// ---------------- problem constants ----------------
#define B_ 32
#define S_ 2048
#define I_ 128
#define H_ 512
#define O_ 128

#define NWAVES 16
#define NTHREADS (NWAVES * 32)

// padded LDS strides (elements) -> 4-bank skew per row, 16B aligned rows
#define HS 520   // bf16 H-sized rows: 1040 B/row
#define XS 136   // bf16 I-sized rows: 272 B/row
#define ZS 520   // f32 z rows

// LDS layout (bytes)
#define SM_ZBUF 0
#define SM_H0A  66560            // 32*520*4
#define SM_H1A  (SM_H0A + 33280) // 32*520*2
#define SM_RHA  (SM_H1A + 33280)
#define SM_XA   (SM_RHA + 33280)
#define SMEM_BYTES (SM_XA + 8704) // 175104 total

// bf16 weight copies in workspace (element offsets)
#define OFF_WX0 0u
#define OFF_WH0 196608u   // 1536*128
#define OFF_WX1 983040u   // + 1536*512
#define OFF_WH1 1769472u
#define OFF_WHY 2555904u
#define W_ELEMS 2621440u

typedef __bf16 v16bf __attribute__((ext_vector_type(16)));
typedef float  v8f   __attribute__((ext_vector_type(8)));

// ---- load 16 contiguous bf16 (32B) ----
__device__ __forceinline__ v16bf load16(const __bf16* p) {
  union { uint4 u[2]; v16bf v; } t;
  t.u[0] = ((const uint4*)p)[0];
  t.u[1] = ((const uint4*)p)[1];
  return t.v;
}

// ---- A operand from LDS: 16x32 bf16 tile, rows rt..rt+15, K base k ----
// lane<16 holds row (lane), K {k..k+7, k+16..k+23}; lane>=16 holds K {k+8..k+15, k+24..k+31}
__device__ __forceinline__ v16bf load_a(const __bf16* A, int as, int rt, int k, int lane) {
  const int row = rt + (lane & 15);
  const int hi  = lane >> 4;
  const __bf16* base = A + row * as + k + 8 * hi;
  union { uint4 u[2]; v16bf v; } t;
  t.u[0] = *(const uint4*)(base);
  t.u[1] = *(const uint4*)(base + 16);
  return t.v;
}

// GEMM column tile: acc(b, j0+n) += A(b, :) * W(j0+n, :)^T for both 16-row tiles.
// B operand: lane n = column j0+n; lanes 0-15 K {k..k+7, k+16..k+23}, lanes 16-31 K {k+8..k+15, k+24..k+31}.
__device__ __forceinline__ void mm_tile(v8f& acc0, v8f& acc1,
                                        const __bf16* A, int as,
                                        const __bf16* __restrict__ W, int ldk,
                                        int j0, int K, int lane) {
  const int n  = lane & 15;
  const int hi = lane >> 4;
  const __bf16* wrow = W + (size_t)(j0 + n) * ldk + 8 * hi;
  for (int k = 0; k < K; k += 32) {
    v16bf bm;
    {
      union { uint4 u[2]; v16bf v; } t;
      t.u[0] = *(const uint4*)(wrow + k);
      t.u[1] = *(const uint4*)(wrow + k + 16);
      bm = t.v;
    }
    v16bf a0 = load_a(A, as, 0, k, lane);
    v16bf a1 = load_a(A, as, 16, k, lane);
    acc0 = __builtin_amdgcn_wmma_f32_16x16x32_bf16(false, a0, false, bm, (short)0, acc0, false, false);
    acc1 = __builtin_amdgcn_wmma_f32_16x16x32_bf16(false, a1, false, bm, (short)0, acc1, false, false);
  }
}

__device__ __forceinline__ float sigf(float x) { return 1.0f / (1.0f + __expf(-x)); }

// ---------------- weight fp32 -> bf16 conversion ----------------
__global__ void gru_convert_weights(const float* __restrict__ Wx0, const float* __restrict__ Wh0,
                                    const float* __restrict__ Wx1, const float* __restrict__ Wh1,
                                    const float* __restrict__ Why, __bf16* __restrict__ dst) {
  unsigned i = blockIdx.x * blockDim.x + threadIdx.x;
  if (i >= W_ELEMS) return;
  float v;
  if      (i < OFF_WH0) v = Wx0[i - OFF_WX0];
  else if (i < OFF_WX1) v = Wh0[i - OFF_WH0];
  else if (i < OFF_WH1) v = Wx1[i - OFF_WX1];
  else if (i < OFF_WHY) v = Wh1[i - OFF_WH1];
  else                  v = Why[i - OFF_WHY];
  dst[i] = (__bf16)v;
}

// ---------------- persistent GRU scan (single WGP, 16 waves) ----------------
__global__ void __launch_bounds__(NTHREADS)
gru_scan_kernel(const float* __restrict__ x_in,
                const float* __restrict__ h0_in,
                const float* __restrict__ bh0,
                const float* __restrict__ bh1,
                const float* __restrict__ bhy,
                const __bf16* __restrict__ Wb,
                float* __restrict__ out) {
  extern __shared__ char smem_raw[];
  float*  zbuf = (float*)(smem_raw + SM_ZBUF);
  __bf16* h0A  = (__bf16*)(smem_raw + SM_H0A);   // layer0 state (also s0 for layer1)
  __bf16* h1A  = (__bf16*)(smem_raw + SM_H1A);   // layer1 state (also s1 for output)
  __bf16* rhA  = (__bf16*)(smem_raw + SM_RHA);   // r * h
  __bf16* xA   = (__bf16*)(smem_raw + SM_XA);    // staged x_t

  const __bf16* Wx0b = Wb + OFF_WX0;
  const __bf16* Wh0b = Wb + OFF_WH0;
  const __bf16* Wx1b = Wb + OFF_WX1;
  const __bf16* Wh1b = Wb + OFF_WH1;
  const __bf16* Whyb = Wb + OFF_WHY;

  const int tid  = threadIdx.x;
  const int wave = tid >> 5;
  const int lane = tid & 31;
  const int n    = lane & 15;
  const int hi   = lane >> 4;

  // initial hidden state -> LDS (bf16)
  for (int idx = tid; idx < B_ * H_; idx += NTHREADS) {
    int b = idx >> 9, j = idx & (H_ - 1);
    h0A[b * HS + j] = (__bf16)h0_in[(b * 2 + 0) * H_ + j];
    h1A[b * HS + j] = (__bf16)h0_in[(b * 2 + 1) * H_ + j];
  }
  __syncthreads();

  for (int s = 0; s < S_; ++s) {
    // stage x_t (fp32 -> bf16) + prefetch next timestep
    for (int idx = tid; idx < B_ * I_; idx += NTHREADS) {
      int b = idx >> 7, i = idx & (I_ - 1);
      xA[b * XS + i] = (__bf16)x_in[((size_t)b * S_ + s) * I_ + i];
    }
    if (s + 1 < S_ && tid < B_)
      __builtin_prefetch(&x_in[((size_t)tid * S_ + (s + 1)) * I_], 0, 1);
    __syncthreads();

    // ---- Phase A: layer0 z and r ----
    for (int t = wave; t < 32; t += NWAVES) {
      const int j0 = 16 * t;
      { // z tile
        v8f a0 = {}, a1 = {};
        mm_tile(a0, a1, xA, XS, Wx0b, I_, j0, I_, lane);
        mm_tile(a0, a1, h0A, HS, Wh0b, H_, j0, H_, lane);
        const float bias = bh0[j0 + n];
        #pragma unroll
        for (int r = 0; r < 8; ++r) {
          int b0 = 8 * hi + r;
          zbuf[b0 * ZS + j0 + n]        = sigf(a0[r] + bias);
          zbuf[(b0 + 16) * ZS + j0 + n] = sigf(a1[r] + bias);
        }
      }
      { // r tile -> rh = r * h
        const int jr = H_ + j0;
        v8f a0 = {}, a1 = {};
        mm_tile(a0, a1, xA, XS, Wx0b, I_, jr, I_, lane);
        mm_tile(a0, a1, h0A, HS, Wh0b, H_, jr, H_, lane);
        const float bias = bh0[jr + n];
        #pragma unroll
        for (int r = 0; r < 8; ++r) {
          #pragma unroll
          for (int half = 0; half < 2; ++half) {
            int b0 = 8 * hi + r + 16 * half;
            float rv = sigf((half ? a1 : a0)[r] + bias);
            rhA[b0 * HS + j0 + n] = (__bf16)(rv * (float)h0A[b0 * HS + j0 + n]);
          }
        }
      }
    }
    __syncthreads();

    // ---- Phase B: layer0 g, state update ----
    for (int t = wave; t < 32; t += NWAVES) {
      const int j0 = 16 * t, jg = 2 * H_ + j0;
      v8f a0 = {}, a1 = {};
      mm_tile(a0, a1, xA, XS, Wx0b, I_, jg, I_, lane);
      mm_tile(a0, a1, rhA, HS, Wh0b, H_, jg, H_, lane);
      const float bias = bh0[jg + n];
      #pragma unroll
      for (int r = 0; r < 8; ++r) {
        #pragma unroll
        for (int half = 0; half < 2; ++half) {
          int b0 = 8 * hi + r + 16 * half;
          float g = tanhf((half ? a1 : a0)[r] + bias);
          float z = zbuf[b0 * ZS + j0 + n];
          float h = (float)h0A[b0 * HS + j0 + n];
          h0A[b0 * HS + j0 + n] = (__bf16)(z * h + (1.0f - z) * g);
        }
      }
    }
    __syncthreads();

    // ---- Phase C: layer1 z and r (input s0 = h0A) ----
    for (int t = wave; t < 32; t += NWAVES) {
      const int j0 = 16 * t;
      { // z tile
        v8f a0 = {}, a1 = {};
        mm_tile(a0, a1, h0A, HS, Wx1b, H_, j0, H_, lane);
        mm_tile(a0, a1, h1A, HS, Wh1b, H_, j0, H_, lane);
        const float bias = bh1[j0 + n];
        #pragma unroll
        for (int r = 0; r < 8; ++r) {
          int b0 = 8 * hi + r;
          zbuf[b0 * ZS + j0 + n]        = sigf(a0[r] + bias);
          zbuf[(b0 + 16) * ZS + j0 + n] = sigf(a1[r] + bias);
        }
      }
      { // r tile
        const int jr = H_ + j0;
        v8f a0 = {}, a1 = {};
        mm_tile(a0, a1, h0A, HS, Wx1b, H_, jr, H_, lane);
        mm_tile(a0, a1, h1A, HS, Wh1b, H_, jr, H_, lane);
        const float bias = bh1[jr + n];
        #pragma unroll
        for (int r = 0; r < 8; ++r) {
          #pragma unroll
          for (int half = 0; half < 2; ++half) {
            int b0 = 8 * hi + r + 16 * half;
            float rv = sigf((half ? a1 : a0)[r] + bias);
            rhA[b0 * HS + j0 + n] = (__bf16)(rv * (float)h1A[b0 * HS + j0 + n]);
          }
        }
      }
    }
    __syncthreads();

    // ---- Phase D: layer1 g, state update ----
    for (int t = wave; t < 32; t += NWAVES) {
      const int j0 = 16 * t, jg = 2 * H_ + j0;
      v8f a0 = {}, a1 = {};
      mm_tile(a0, a1, h0A, HS, Wx1b, H_, jg, H_, lane);
      mm_tile(a0, a1, rhA, HS, Wh1b, H_, jg, H_, lane);
      const float bias = bh1[jg + n];
      #pragma unroll
      for (int r = 0; r < 8; ++r) {
        #pragma unroll
        for (int half = 0; half < 2; ++half) {
          int b0 = 8 * hi + r + 16 * half;
          float g = tanhf((half ? a1 : a0)[r] + bias);
          float z = zbuf[b0 * ZS + j0 + n];
          float h = (float)h1A[b0 * HS + j0 + n];
          h1A[b0 * HS + j0 + n] = (__bf16)(z * h + (1.0f - z) * g);
        }
      }
    }
    __syncthreads();

    // ---- Phase E: output projection y = s1 @ Why^T + bhy ----
    for (int t = wave; t < O_ / 16; t += NWAVES) {
      const int j0 = 16 * t;
      v8f a0 = {}, a1 = {};
      mm_tile(a0, a1, h1A, HS, Whyb, H_, j0, H_, lane);
      const float bias = bhy[j0 + n];
      #pragma unroll
      for (int r = 0; r < 8; ++r) {
        int b0 = 8 * hi + r;
        out[((size_t)b0 * S_ + s) * O_ + j0 + n]        = a0[r] + bias;
        out[((size_t)(b0 + 16) * S_ + s) * O_ + j0 + n] = a1[r] + bias;
      }
    }
    __syncthreads();
  }

  // final hidden state (B, L=2, H) appended after layer_output
  float* hs = out + (size_t)B_ * S_ * O_;
  for (int idx = tid; idx < B_ * H_; idx += NTHREADS) {
    int b = idx >> 9, j = idx & (H_ - 1);
    hs[(b * 2 + 0) * H_ + j] = (float)h0A[b * HS + j];
    hs[(b * 2 + 1) * H_ + j] = (float)h1A[b * HS + j];
  }
}

extern "C" void kernel_launch(void* const* d_in, const int* in_sizes, int n_in,
                              void* d_out, int out_size, void* d_ws, size_t ws_size,
                              hipStream_t stream) {
  (void)in_sizes; (void)n_in; (void)out_size; (void)ws_size;
  const float* input = (const float*)d_in[0];
  const float* h0    = (const float*)d_in[1];
  const float* Wx0   = (const float*)d_in[2];
  const float* Wh0   = (const float*)d_in[3];
  const float* bh0   = (const float*)d_in[4];
  const float* Wx1   = (const float*)d_in[5];
  const float* Wh1   = (const float*)d_in[6];
  const float* bh1   = (const float*)d_in[7];
  const float* Why   = (const float*)d_in[8];
  const float* bhy   = (const float*)d_in[9];
  float*  out = (float*)d_out;
  __bf16* Wb  = (__bf16*)d_ws;

  gru_convert_weights<<<(W_ELEMS + 255) / 256, 256, 0, stream>>>(Wx0, Wh0, Wx1, Wh1, Why, Wb);

  (void)hipFuncSetAttribute(reinterpret_cast<const void*>(gru_scan_kernel),
                            hipFuncAttributeMaxDynamicSharedMemorySize, SMEM_BYTES);
  gru_scan_kernel<<<1, NTHREADS, SMEM_BYTES, stream>>>(input, h0, bh0, bh1, bhy, Wb, out);
}
